// LAS_56453050138644
// MI455X (gfx1250) — matrix operand order repaired and checked by reading, not compiled, not run on hardware
//
#include <hip/hip_runtime.h>

// ---------------- constants ----------------
#define B_  64
#define T_  2048
#define D_  257
#define H_  64
#define G4_ 256          // 4*H
#define KD_ 288          // D padded to mult of 32
#define DEC_IN_ 321
#define KDEC_ 352        // DEC_IN padded
#define KPI_ 608         // 578 padded
#define NCLS_ 27
#define MAXDEC_ 96

typedef __attribute__((ext_vector_type(16))) _Float16 v16h;
typedef __attribute__((ext_vector_type(8)))  float    v8f;

__device__ __forceinline__ float sigmf(float x) { return 1.f / (1.f + __expf(-x)); }

// ---- WMMA fragment loaders (contiguous k order, same in A and B, wave32) ----
// A: 16x32 f16, lane L: m = L&15, hlf=L>>4; k=hlf*16+e
__device__ __forceinline__ v16h load_a_frag(const _Float16* A, int lda, int mbase, int k0, int lane) {
    const int m   = mbase + (lane & 15);
    const int hlf = lane >> 4;
    const _Float16* row = A + (long long)m * lda + k0;
    v16h a;
#pragma unroll
    for (int e = 0; e < 16; ++e) a[e] = row[hlf * 16 + e];
    return a;
}
// B: 32x16 f16 = W^T tile; W is row-major N x K. lane: n = L&15; hlf selects k half.
__device__ __forceinline__ v16h load_b_frag(const _Float16* W, int ldw, int nbase, int k0, int lane) {
    const int n   = nbase + (lane & 15);
    const int hlf = lane >> 4;
    const _Float16* row = W + (long long)n * ldw + k0 + hlf * 16;
    v16h b;
#pragma unroll
    for (int e = 0; e < 16; ++e) b[e] = row[e];
    return b;
}

// ---- in-workgroup GEMM: C[64,N](f32 LDS) = A[64,K](f16 LDS) @ W[N,K]^T(f16 global) (+bias)(+=C)
__device__ __forceinline__ void wg_gemm(const _Float16* A_lds, int lda,
                                        const _Float16* Wg, int ldw,
                                        const float* bias, float* C, int ldc,
                                        int N, int K, bool accumulate) {
    const int tid  = threadIdx.x;
    const int lane = tid & 31;
    const int wave = tid >> 5;
    const int hlf  = lane >> 4;
    const int nl   = lane & 15;
    const int ntiles = (N + 15) >> 4;
    for (int nt = wave; nt < ntiles; nt += 8) {
        const int n = nt * 16 + nl;
        for (int mi = 0; mi < 4; ++mi) {
            v8f acc = {};
            for (int k0 = 0; k0 < K; k0 += 32) {
                v16h a = load_a_frag(A_lds, lda, mi * 16, k0, lane);
                v16h b;
                if (n < N) {
                    const _Float16* wr = Wg + (long long)n * ldw + k0 + hlf * 16;
#pragma unroll
                    for (int e = 0; e < 16; ++e) b[e] = wr[e];
                    if (k0 + 32 < K) __builtin_prefetch(wr + 32, 0, 1);  // global_prefetch_b8
                } else {
#pragma unroll
                    for (int e = 0; e < 16; ++e) b[e] = (_Float16)0.f;
                }
                acc = __builtin_amdgcn_wmma_f32_16x16x32_f16(false, a, false, b, (short)0, acc, false, false);
            }
            if (n < N) {
#pragma unroll
                for (int r = 0; r < 8; ++r) {
                    int m = mi * 16 + hlf * 8 + r;
                    float v = acc[r];
                    if (bias) v += bias[n];
                    if (accumulate) v += C[m * ldc + n];
                    C[m * ldc + n] = v;
                }
            }
        }
    }
}

// ---------------- generic tiled WMMA GEMM: out = A@W^T (+bias)(+res)(mask) ----------------
// wg tile 64x64, 256 threads, 8 waves x 2 (16x16) tiles.
// Software-pipelined: double-buffered LDS, one barrier per K-step, global loads for
// tile k+2 issued in registers while WMMA consumes tile k; deeper tiles prefetched.
__global__ void __launch_bounds__(256) k_gemm64(
    const _Float16* __restrict__ A, int lda,
    const _Float16* __restrict__ W, int ldw,
    const float* __restrict__ bias,
    const float* res, int ldres,
    float* out32, int ldo32,
    _Float16* out16, int ldo16,
    int N, int K,
    const int* __restrict__ lens, int Tdim) {
    __shared__ _Float16 As[2][64 * 32];
    __shared__ _Float16 Bs[2][64 * 32];
    const int tid = threadIdx.x;
    const int lane = tid & 31, wave = tid >> 5;
    const int hlf = lane >> 4, nl = lane & 15;
    const long long m0 = (long long)blockIdx.x * 64;
    const int n0blk = blockIdx.y * 64;
    const int lrow = tid >> 2;          // 0..63
    const int lcol = (tid & 3) * 8;     // 0,8,16,24 (halfs)
    const int lofs = lrow * 32 + lcol;
    const int wn = n0blk + lrow;
    v8f zero = {};
    v8f acc[2] = {zero, zero};

    const _Float16* arow = A + (m0 + lrow) * lda + lcol;
    const _Float16* wrow = W + (long long)wn * ldw + lcol;
    uint4 zed; zed.x = zed.y = zed.z = zed.w = 0u;

    const int nk = K >> 5;
    uint4 rA = *(const uint4*)(arow);
    uint4 rB = (wn < N) ? *(const uint4*)(wrow) : zed;
    *(uint4*)&As[0][lofs] = rA;
    *(uint4*)&Bs[0][lofs] = rB;
    if (nk > 1) {
        rA = *(const uint4*)(arow + 32);
        rB = (wn < N) ? *(const uint4*)(wrow + 32) : zed;
    }
    __syncthreads();
    for (int kk = 0; kk < nk; ++kk) {
        const int buf = kk & 1;
        if (kk + 1 < nk) {
            *(uint4*)&As[buf ^ 1][lofs] = rA;
            *(uint4*)&Bs[buf ^ 1][lofs] = rB;
            if (kk + 2 < nk) {
                rA = *(const uint4*)(arow + (kk + 2) * 32);
                rB = (wn < N) ? *(const uint4*)(wrow + (kk + 2) * 32) : zed;
            }
            if (kk + 4 < nk) {
                __builtin_prefetch(arow + (kk + 4) * 32, 0, 1);      // global_prefetch_b8
                if (wn < N) __builtin_prefetch(wrow + (kk + 4) * 32, 0, 1);
            }
        }
#pragma unroll
        for (int tt = 0; tt < 2; ++tt) {
            int tile = wave * 2 + tt;
            int mi = tile >> 2, ni = tile & 3;
            v16h a = load_a_frag(As[buf], 32, mi * 16, 0, lane);
            v16h b = load_b_frag(Bs[buf], 32, ni * 16, 0, lane);
            acc[tt] = __builtin_amdgcn_wmma_f32_16x16x32_f16(false, a, false, b, (short)0, acc[tt], false, false);
        }
        __syncthreads();
    }
#pragma unroll
    for (int tt = 0; tt < 2; ++tt) {
        int tile = wave * 2 + tt;
        int mi = tile >> 2, ni = tile & 3;
        int n = n0blk + ni * 16 + nl;
        if (n >= N) continue;
#pragma unroll
        for (int r = 0; r < 8; ++r) {
            long long m = m0 + mi * 16 + hlf * 8 + r;
            float v = acc[tt][r];
            if (bias) v += bias[n];
            if (res)  v += res[m * ldres + n];
            if (lens) {
                int bb = (int)(m / Tdim), t = (int)(m % Tdim);
                if (t >= lens[bb]) v = 0.f;
            }
            if (out32) out32[m * (long long)ldo32 + n] = v;
            if (out16) out16[m * (long long)ldo16 + n] = (_Float16)v;
        }
    }
}

// ---------------- persistent recurrent LSTM (1 workgroup, loops over T) ----------------
__global__ void __launch_bounds__(256) k_lstm_seq(
    const float* __restrict__ gates,       // [B,T,256], bias already added
    const _Float16* __restrict__ Whh16,    // [256,64]
    const int* __restrict__ lens,
    _Float16* __restrict__ hout, int rowStride, int Tdim) {
    extern __shared__ char smem[];
    _Float16* Whh_s = (_Float16*)smem;             // 32768 B
    _Float16* h16   = (_Float16*)(smem + 32768);   //  8192 B
    float*    cbuf  = (float*)(smem + 40960);      // 16384 B
    float*    gbuf  = (float*)(smem + 57344);      // 65536 B
    const int tid = threadIdx.x;
    const int lane = tid & 31, wave = tid >> 5, hlf = lane >> 4, nl = lane & 15;
    for (int i = tid; i < 256 * 64; i += 256) Whh_s[i] = Whh16[i];
    for (int i = tid; i < 64 * 64; i += 256) { h16[i] = (_Float16)0.f; cbuf[i] = 0.f; }
    __syncthreads();
    for (int t = 0; t < Tdim; ++t) {
        for (int it = 0; it < 8; ++it) {
            int tile = wave + 8 * it;           // 0..63
            int mi = tile >> 4, ni = tile & 15;
            v8f acc = {};
#pragma unroll
            for (int k0 = 0; k0 < 64; k0 += 32) {
                v16h a = load_a_frag(h16, 64, mi * 16, k0, lane);
                v16h b = load_b_frag(Whh_s, 64, ni * 16, k0, lane);
                acc = __builtin_amdgcn_wmma_f32_16x16x32_f16(false, a, false, b, (short)0, acc, false, false);
            }
            int n = ni * 16 + nl;
            // prefetch next timestep's gate row while this step's math runs
            if (t + 1 < Tdim)
                __builtin_prefetch(&gates[((long long)(mi * 16 + hlf * 8) * Tdim + t + 1) * 256 + n], 0, 1);
#pragma unroll
            for (int r = 0; r < 8; ++r) {
                int m = mi * 16 + hlf * 8 + r;
                gbuf[m * 256 + n] = acc[r] + gates[((long long)m * Tdim + t) * 256 + n];
            }
        }
        __syncthreads();
        for (int i = tid; i < 64 * 64; i += 256) {
            int bb = i >> 6, hh = i & 63;
            float gi = gbuf[bb * 256 + hh];
            float gf = gbuf[bb * 256 + 64 + hh];
            float gg = gbuf[bb * 256 + 128 + hh];
            float go = gbuf[bb * 256 + 192 + hh];
            float c2 = sigmf(gf) * cbuf[i] + sigmf(gi) * tanhf(gg);
            float h2 = sigmf(go) * tanhf(c2);
            float outv = 0.f;
            if (t < lens[bb]) { cbuf[i] = c2; h16[i] = (_Float16)h2; outv = h2; }
            hout[((long long)bb * Tdim + t) * rowStride + hh] = (_Float16)outv;
        }
        __syncthreads();
    }
}

// ---------------- data movement / pointwise kernels ----------------
__global__ void k_pad(const float* __restrict__ in, const int* __restrict__ lens,
                      float* __restrict__ xf32, _Float16* __restrict__ xpad) {
    long long i = (long long)blockIdx.x * 256 + threadIdx.x;
    if (i >= (long long)B_ * T_ * KD_) return;
    int k = (int)(i % KD_);
    long long bt = i / KD_;
    int t = (int)(bt % T_), bb = (int)(bt / T_);
    float v = 0.f;
    if (k < D_ && t < lens[bb]) v = in[bt * D_ + k];
    xpad[i] = (_Float16)v;
    if (k < D_) xf32[bt * D_ + k] = v;
}

__global__ void k_revpad(const float* __restrict__ xf32, const int* __restrict__ lens,
                         _Float16* __restrict__ xr) {
    long long i = (long long)blockIdx.x * 256 + threadIdx.x;
    if (i >= (long long)B_ * T_ * KD_) return;
    int k = (int)(i % KD_);
    long long bt = i / KD_;
    int s = (int)(bt % T_), bb = (int)(bt / T_);
    int L = lens[bb];
    float v = 0.f;
    if (k < D_ && s < L) v = xf32[((long long)bb * T_ + (L - 1 - s)) * D_ + k];
    xr[i] = (_Float16)v;
}

__global__ void k_unrev(const _Float16* __restrict__ hrev, const int* __restrict__ lens,
                        _Float16* __restrict__ rnncat) {
    long long i = (long long)blockIdx.x * 256 + threadIdx.x;
    if (i >= (long long)B_ * T_ * 64) return;
    int hh = (int)(i & 63);
    long long bt = i >> 6;
    int t = (int)(bt % T_), bb = (int)(bt / T_);
    int L = lens[bb];
    _Float16 v = (_Float16)0.f;
    if (t < L) v = hrev[((long long)bb * T_ + (L - 1 - t)) * 64 + hh];
    rnncat[bt * 128 + 64 + hh] = v;
}

__global__ void k_cvt(const float* __restrict__ src, _Float16* __restrict__ dst,
                      int rows, int cols, int ds) {
    long long i = (long long)blockIdx.x * 256 + threadIdx.x;
    if (i >= (long long)rows * ds) return;
    int r = (int)(i / ds), c = (int)(i % ds);
    dst[i] = (c < cols) ? (_Float16)src[(long long)r * cols + c] : (_Float16)0.f;
}

__global__ void k_zero(float* p, long long n) {
    long long i = (long long)blockIdx.x * 256 + threadIdx.x;
    if (i < n) p[i] = 0.f;
}

// ---------------- decoder kernels ----------------
__global__ void __launch_bounds__(256) k_dec_cell(
    const float* __restrict__ E, const int* __restrict__ tgt,
    const float* __restrict__ ctx,
    float* h1, float* c1, float* h2, float* c2,
    float* out2g, float* qg,
    const _Float16* Wih0, const _Float16* Whh0, const _Float16* Wp0,
    const _Float16* Wih1, const _Float16* Whh1, const _Float16* Wp1d,
    const _Float16* Wfc,
    const float* b0, const float* bp0,
    const float* b1, const float* bp1d,
    const float* bfc, int step) {
    extern __shared__ char smem[];
    _Float16* Abuf = (_Float16*)smem;               // 64*352*2 = 45056
    _Float16* Hbuf = (_Float16*)(smem + 45056);     // 64*64*2  = 8192
    float*    Cbuf = (float*)(smem + 53248);        // 64*336*4 = 86016
    float*    Obuf = (float*)(smem + 139264);       // 64*336*4 = 86016
    const int tid = threadIdx.x;
    // dec_in = [E[prev_label], ctx], pad to 352
    for (int i = tid; i < 64 * KDEC_; i += 256) {
        int bb = i / KDEC_, k = i % KDEC_;
        float v = 0.f;
        if (k < 64) {
            int lab = (step == 0) ? NCLS_ : tgt[bb * MAXDEC_ + step - 1];
            v = E[lab * 64 + k];
        } else if (k < DEC_IN_) v = ctx[bb * D_ + (k - 64)];
        Abuf[i] = (_Float16)v;
    }
    for (int i = tid; i < 4096; i += 256) Hbuf[i] = (_Float16)h1[i];
    __syncthreads();
    wg_gemm(Abuf, KDEC_, Wih0, KDEC_, b0, Cbuf, 336, 256, KDEC_, false);
    __syncthreads();
    wg_gemm(Hbuf, 64, Whh0, 64, nullptr, Cbuf, 336, 256, 64, true);
    __syncthreads();
    for (int i = tid; i < 4096; i += 256) {   // LSTM cell 1
        int bb = i >> 6, hh = i & 63;
        float gi = Cbuf[bb * 336 + hh], gf = Cbuf[bb * 336 + 64 + hh];
        float gg = Cbuf[bb * 336 + 128 + hh], go = Cbuf[bb * 336 + 192 + hh];
        float cc = sigmf(gf) * c1[i] + sigmf(gi) * tanhf(gg);
        float hv = sigmf(go) * tanhf(cc);
        c1[i] = cc; h1[i] = hv; Hbuf[i] = (_Float16)hv;
    }
    __syncthreads();
    wg_gemm(Hbuf, 64, Wp0, 64, bp0, Obuf, 336, DEC_IN_, 64, false);   // h1@Wp0^T + bp0
    __syncthreads();
    for (int i = tid; i < 64 * DEC_IN_; i += 256) {                   // + dec_in
        int bb = i / DEC_IN_, n = i % DEC_IN_;
        Obuf[bb * 336 + n] += (float)Abuf[bb * KDEC_ + n];
    }
    __syncthreads();
    for (int i = tid; i < 64 * KDEC_; i += 256) {                     // Abuf = out1 (f16, padded)
        int bb = i / KDEC_, k = i % KDEC_;
        Abuf[i] = (k < DEC_IN_) ? (_Float16)Obuf[bb * 336 + k] : (_Float16)0.f;
    }
    for (int i = tid; i < 4096; i += 256) Hbuf[i] = (_Float16)h2[i];
    __syncthreads();
    wg_gemm(Abuf, KDEC_, Wih1, KDEC_, b1, Cbuf, 336, 256, KDEC_, false);
    __syncthreads();
    wg_gemm(Hbuf, 64, Whh1, 64, nullptr, Cbuf, 336, 256, 64, true);
    __syncthreads();
    for (int i = tid; i < 4096; i += 256) {   // LSTM cell 2
        int bb = i >> 6, hh = i & 63;
        float gi = Cbuf[bb * 336 + hh], gf = Cbuf[bb * 336 + 64 + hh];
        float gg = Cbuf[bb * 336 + 128 + hh], go = Cbuf[bb * 336 + 192 + hh];
        float cc = sigmf(gf) * c2[i] + sigmf(gi) * tanhf(gg);
        float hv = sigmf(go) * tanhf(cc);
        c2[i] = cc; h2[i] = hv; Hbuf[i] = (_Float16)hv;
    }
    __syncthreads();
    wg_gemm(Hbuf, 64, Wp1d, 64, bp1d, Cbuf, 336, DEC_IN_, 64, false); // h2@Wp1^T + bp1
    __syncthreads();
    for (int i = tid; i < 64 * DEC_IN_; i += 256) {                   // out2 = out1 + ...
        int bb = i / DEC_IN_, n = i % DEC_IN_;
        float v = Obuf[bb * 336 + n] + Cbuf[bb * 336 + n];
        Obuf[bb * 336 + n] = v;
        out2g[bb * 336 + n] = v;
    }
    __syncthreads();
    for (int i = tid; i < 64 * KDEC_; i += 256) {
        int bb = i / KDEC_, k = i % KDEC_;
        Abuf[i] = (k < DEC_IN_) ? (_Float16)Obuf[bb * 336 + k] : (_Float16)0.f;
    }
    __syncthreads();
    wg_gemm(Abuf, KDEC_, Wfc, KDEC_, bfc, Cbuf, 336, D_, KDEC_, false); // q
    __syncthreads();
    for (int i = tid; i < 64 * D_; i += 256) {
        int bb = i / D_, n = i % D_;
        qg[i] = Cbuf[bb * 336 + n];
    }
}

__global__ void k_sim(const float* __restrict__ enc, const float* __restrict__ q,
                      float* __restrict__ sim) {
    int bb = blockIdx.x;
    int wave = threadIdx.x >> 5, lane = threadIdx.x & 31;
    int t = blockIdx.y * 8 + wave;
    const float* er = enc + ((long long)bb * T_ + t) * D_;
    const float* qr = q + bb * D_;
    float s = 0.f;
    for (int d = lane; d < D_; d += 32) s += er[d] * qr[d];
#pragma unroll
    for (int off = 16; off > 0; off >>= 1) s += __shfl_down(s, off, 32);
    if (lane == 0) sim[(long long)bb * T_ + t] = s;
}

__global__ void k_softmax(float* __restrict__ sim, const int* __restrict__ lens) {
    __shared__ float red[256];
    int bb = blockIdx.x, tid = threadIdx.x, L = lens[bb];
    float* row = sim + (long long)bb * T_;
    float mx = -1e30f;
    for (int t = tid; t < L; t += 256) mx = fmaxf(mx, row[t]);
    red[tid] = mx; __syncthreads();
    for (int s = 128; s > 0; s >>= 1) { if (tid < s) red[tid] = fmaxf(red[tid], red[tid + s]); __syncthreads(); }
    mx = red[0]; __syncthreads();
    float sum = 0.f;
    for (int t = tid; t < L; t += 256) sum += __expf(row[t] - mx);
    red[tid] = sum; __syncthreads();
    for (int s = 128; s > 0; s >>= 1) { if (tid < s) red[tid] += red[tid + s]; __syncthreads(); }
    float inv = 1.f / red[0]; __syncthreads();
    for (int t = tid; t < T_; t += 256) row[t] = (t < L) ? __expf(row[t] - mx) * inv : 0.f;
}

__global__ void k_ctx(const float* __restrict__ enc, const float* __restrict__ attn,
                      float* __restrict__ ctx) {
    int bb = blockIdx.x, tid = threadIdx.x;
    for (int d = tid; d < D_; d += 256) {
        float a = 0.f;
        for (int t = 0; t < T_; ++t)
            a += attn[(long long)bb * T_ + t] * enc[((long long)bb * T_ + t) * D_ + d];
        ctx[bb * D_ + d] = a;
    }
}

__global__ void __launch_bounds__(256) k_dec_out(
    const float* __restrict__ out2g, const float* __restrict__ ctx,
    const _Float16* Wp1, const _Float16* Wp2,
    const float* bp1, const float* bp2,
    float* __restrict__ dout, int step) {
    extern __shared__ char smem[];
    _Float16* Abuf = (_Float16*)smem;              // 64*608*2 = 77824
    _Float16* Hbuf = (_Float16*)(smem + 77824);    // 8192
    float*    Cbuf = (float*)(smem + 86016);       // 64*64*4 = 16384
    float*    Pbuf = (float*)(smem + 102400);      // 64*32*4 = 8192
    const int tid = threadIdx.x;
    for (int i = tid; i < 64 * KPI_; i += 256) {
        int bb = i / KPI_, k = i % KPI_;
        float v = 0.f;
        if (k < DEC_IN_) v = out2g[bb * 336 + k];
        else if (k < DEC_IN_ + D_) v = ctx[bb * D_ + (k - DEC_IN_)];
        Abuf[i] = (_Float16)v;
    }
    __syncthreads();
    wg_gemm(Abuf, KPI_, Wp1, KPI_, bp1, Cbuf, 64, 64, KPI_, false);
    __syncthreads();
    for (int i = tid; i < 4096; i += 256) Hbuf[i] = (_Float16)tanhf(Cbuf[i]);
    __syncthreads();
    wg_gemm(Hbuf, 64, Wp2, 64, bp2, Pbuf, 32, NCLS_, 64, false);
    __syncthreads();
    if (tid < 64) {
        int bb = tid;
        float best = -1e30f; int bi = 0;
        for (int j = 0; j < NCLS_; ++j) {
            float v = Pbuf[bb * 32 + j];
            dout[((long long)bb * MAXDEC_ + step) * NCLS_ + j] = v;
            if (v > best) { best = v; bi = j; }
        }
        dout[(long long)B_ * MAXDEC_ * NCLS_ + bb * MAXDEC_ + step] = (float)bi;
    }
}

// ---------------- host ----------------
extern "C" void kernel_launch(void* const* d_in, const int* in_sizes, int n_in,
                              void* d_out, int out_size, void* d_ws, size_t ws_size,
                              hipStream_t stream) {
    (void)in_sizes; (void)n_in; (void)out_size; (void)ws_size;
    // JAX pytree flatten order: top-level dict insertion order, nested dicts by sorted keys.
    const float* inFeat = (const float*)d_in[0];
    const int*   lens   = (const int*)d_in[1];
    const int*   tgt    = (const int*)d_in[2];
    const float* Wfc = (const float*)d_in[3];
    const float* Wp1 = (const float*)d_in[4];
    const float* Wp2 = (const float*)d_in[5];
    const float* bfc = (const float*)d_in[6];
    const float* bp1 = (const float*)d_in[7];
    const float* bp2 = (const float*)d_in[8];
    const float* dWhh[2] = {(const float*)d_in[9],  (const float*)d_in[14]};
    const float* dWih[2] = {(const float*)d_in[10], (const float*)d_in[15]};
    const float* dWp[2]  = {(const float*)d_in[11], (const float*)d_in[16]};
    const float* db[2]   = {(const float*)d_in[12], (const float*)d_in[17]};
    const float* dbp[2]  = {(const float*)d_in[13], (const float*)d_in[18]};
    const float* Emb = (const float*)d_in[19];
    const float* eWhh_b[2] = {(const float*)d_in[20], (const float*)d_in[28]};
    const float* eWhh_f[2] = {(const float*)d_in[21], (const float*)d_in[29]};
    const float* eWih_b[2] = {(const float*)d_in[22], (const float*)d_in[30]};
    const float* eWih_f[2] = {(const float*)d_in[23], (const float*)d_in[31]};
    const float* eWp[2]    = {(const float*)d_in[24], (const float*)d_in[32]};
    const float* eb_b[2]   = {(const float*)d_in[25], (const float*)d_in[33]};
    const float* eb_f[2]   = {(const float*)d_in[26], (const float*)d_in[34]};
    const float* ebp[2]    = {(const float*)d_in[27], (const float*)d_in[35]};

    char* ws = (char*)d_ws;
    size_t off = 0;
    auto alloc = [&](size_t bytes) -> char* {
        char* p = ws + off;
        off = (off + bytes + 255) & ~(size_t)255;
        return p;
    };
    const long long MT = (long long)B_ * T_;
    float*     xf32   = (float*)alloc((size_t)MT * D_ * 4);
    _Float16*  xpad   = (_Float16*)alloc((size_t)MT * KD_ * 2);
    _Float16*  xrpad  = (_Float16*)alloc((size_t)MT * KD_ * 2);
    float*     gatesF = (float*)alloc((size_t)MT * G4_ * 4);
    float*     gatesB = (float*)alloc((size_t)MT * G4_ * 4);
    _Float16*  rnncat = (_Float16*)alloc((size_t)MT * 128 * 2);
    _Float16*  hrev   = (_Float16*)alloc((size_t)MT * 64 * 2);
    float*     simb   = (float*)alloc((size_t)MT * 4);
    float*     ctx    = (float*)alloc((size_t)B_ * D_ * 4);
    float*     qv     = (float*)alloc((size_t)B_ * D_ * 4);
    float*     out2g  = (float*)alloc((size_t)B_ * 336 * 4);
    float*     h1 = (float*)alloc(4096 * 4);
    float*     c1 = (float*)alloc(4096 * 4);
    float*     h2 = (float*)alloc(4096 * 4);
    float*     c2 = (float*)alloc(4096 * 4);
    _Float16 *eWihF16f[2], *eWihF16b[2], *eWhhF16f[2], *eWhhF16b[2], *eWpF16[2];
    _Float16 *dWihF16[2], *dWhhF16[2], *dWpF16[2];
    for (int l = 0; l < 2; ++l) {
        eWihF16f[l] = (_Float16*)alloc(256 * KD_ * 2);
        eWihF16b[l] = (_Float16*)alloc(256 * KD_ * 2);
        eWhhF16f[l] = (_Float16*)alloc(256 * 64 * 2);
        eWhhF16b[l] = (_Float16*)alloc(256 * 64 * 2);
        eWpF16[l]   = (_Float16*)alloc(D_ * 128 * 2);
        dWihF16[l]  = (_Float16*)alloc(256 * KDEC_ * 2);
        dWhhF16[l]  = (_Float16*)alloc(256 * 64 * 2);
        dWpF16[l]   = (_Float16*)alloc(DEC_IN_ * 64 * 2);
    }
    _Float16* WfcF16 = (_Float16*)alloc(D_ * KDEC_ * 2);
    _Float16* Wp1F16 = (_Float16*)alloc(64 * KPI_ * 2);
    _Float16* Wp2F16 = (_Float16*)alloc(NCLS_ * 64 * 2);

    auto cvt = [&](const float* src, _Float16* dst, int rows, int cols, int ds) {
        long long n = (long long)rows * ds;
        k_cvt<<<dim3((unsigned)((n + 255) / 256)), dim3(256), 0, stream>>>(src, dst, rows, cols, ds);
    };
    for (int l = 0; l < 2; ++l) {
        cvt(eWih_f[l], eWihF16f[l], 256, D_, KD_);
        cvt(eWih_b[l], eWihF16b[l], 256, D_, KD_);
        cvt(eWhh_f[l], eWhhF16f[l], 256, 64, 64);
        cvt(eWhh_b[l], eWhhF16b[l], 256, 64, 64);
        cvt(eWp[l],    eWpF16[l],   D_, 128, 128);
        cvt(dWih[l],   dWihF16[l],  256, DEC_IN_, KDEC_);
        cvt(dWhh[l],   dWhhF16[l],  256, 64, 64);
        cvt(dWp[l],    dWpF16[l],   DEC_IN_, 64, 64);
    }
    cvt(Wfc, WfcF16, D_, DEC_IN_, KDEC_);
    cvt(Wp1, Wp1F16, 64, DEC_IN_ + D_, KPI_);
    cvt(Wp2, Wp2F16, NCLS_, 64, 64);

    // ---- encoder ----
    long long nPad = MT * KD_;
    k_pad<<<dim3((unsigned)((nPad + 255) / 256)), dim3(256), 0, stream>>>(inFeat, lens, xf32, xpad);
    for (int l = 0; l < 2; ++l) {
        k_revpad<<<dim3((unsigned)((nPad + 255) / 256)), dim3(256), 0, stream>>>(xf32, lens, xrpad);
        dim3 gGates((unsigned)(MT / 64), 4);
        k_gemm64<<<gGates, dim3(256), 0, stream>>>(xpad, KD_, eWihF16f[l], KD_, eb_f[l],
            nullptr, 0, gatesF, 256, nullptr, 0, 256, KD_, nullptr, 0);
        k_gemm64<<<gGates, dim3(256), 0, stream>>>(xrpad, KD_, eWihF16b[l], KD_, eb_b[l],
            nullptr, 0, gatesB, 256, nullptr, 0, 256, KD_, nullptr, 0);
        k_lstm_seq<<<dim3(1), dim3(256), 122880, stream>>>(gatesF, eWhhF16f[l], lens, rnncat, 128, T_);
        k_lstm_seq<<<dim3(1), dim3(256), 122880, stream>>>(gatesB, eWhhF16b[l], lens, hrev, 64, T_);
        k_unrev<<<dim3((unsigned)((MT * 64 + 255) / 256)), dim3(256), 0, stream>>>(hrev, lens, rnncat);
        dim3 gProj((unsigned)(MT / 64), (D_ + 63) / 64);
        k_gemm64<<<gProj, dim3(256), 0, stream>>>(rnncat, 128, eWpF16[l], 128, ebp[l],
            xf32, D_, xf32, D_, xpad, KD_, D_, 128, lens, T_);
    }
    // xf32 is now enc_output.

    // ---- decoder ----
    k_zero<<<dim3(17), dim3(256), 0, stream>>>(h1, 4096);
    k_zero<<<dim3(17), dim3(256), 0, stream>>>(c1, 4096);
    k_zero<<<dim3(17), dim3(256), 0, stream>>>(h2, 4096);
    k_zero<<<dim3(17), dim3(256), 0, stream>>>(c2, 4096);
    k_zero<<<dim3(65), dim3(256), 0, stream>>>(ctx, (long long)B_ * D_);
    for (int s = 0; s < MAXDEC_; ++s) {
        k_dec_cell<<<dim3(1), dim3(256), 225280, stream>>>(Emb, tgt, ctx, h1, c1, h2, c2, out2g, qv,
            dWihF16[0], dWhhF16[0], dWpF16[0], dWihF16[1], dWhhF16[1], dWpF16[1], WfcF16,
            db[0], dbp[0], db[1], dbp[1], bfc, s);
        k_sim<<<dim3(64, 256), dim3(256), 0, stream>>>(xf32, qv, simb);
        k_softmax<<<dim3(64), dim3(256), 0, stream>>>(simb, lens);
        k_ctx<<<dim3(64), dim3(256), 0, stream>>>(xf32, simb, ctx);
        k_dec_out<<<dim3(1), dim3(256), 110592, stream>>>(out2g, ctx, Wp1F16, Wp2F16, bp1, bp2,
            (float*)d_out, s);
    }
}
